// BidirectionalCrossBlock_10754598109693
// MI455X (gfx1250) — matrix operand order repaired and checked by reading, not compiled, not run on hardware
//
#include <hip/hip_runtime.h>

// ---------------------------------------------------------------------------
// Types for CDNA5 WMMA (wave32): bf16 A/B fragments, f32 C/D accumulators.
// ---------------------------------------------------------------------------
typedef __bf16   v16bf __attribute__((ext_vector_type(16)));
typedef __bf16   v8bf  __attribute__((ext_vector_type(8)));
typedef float    v8f   __attribute__((ext_vector_type(8)));
typedef float    v4f   __attribute__((ext_vector_type(4)));
typedef unsigned u32x4 __attribute__((ext_vector_type(4)));
typedef unsigned u32x8 __attribute__((ext_vector_type(8)));

__device__ __forceinline__ v8f wmma_bf16(v16bf a, v16bf b, v8f c) {
  // D = A(16x32 bf16) x B(32x16 bf16) + C(16x16 f32)
  return __builtin_amdgcn_wmma_f32_16x16x32_bf16(
      /*neg_a=*/false, a, /*neg_b=*/false, b,
      /*c_mod=*/(short)0, c, /*reuse_a=*/false, /*reuse_b=*/false);
}

__device__ __forceinline__ v16bf combine16(v8bf lo, v8bf hi) {
  v16bf r;
#pragma unroll
  for (int i = 0; i < 8; ++i) { r[i] = lo[i]; r[8 + i] = hi[i]; }
  return r;
}

// A fragment (16x32) from row-major [*, ldk] bf16.
// Lanes 0-15: row m=lane, K = {kb..kb+7, kb+16..kb+23}, kb = 8*(lane>=16).
__device__ __forceinline__ v16bf load_afrag(const __bf16* __restrict__ base,
                                            int mrow, int ldk, int k0,
                                            int l15, int hi) {
  const __bf16* p = base + (size_t)mrow * ldk + k0 + hi * 8;
  return combine16(*(const v8bf*)(p), *(const v8bf*)(p + 16));
}

// B fragment (32x16) from row-major W[N,K]: lane col n = lane&15,
// K elements = k0 + 8*(lane>=16) + {0..7, 16..23}  -> two 16-byte chunks.
__device__ __forceinline__ v16bf load_bfrag_rowmajor(const __bf16* __restrict__ W,
                                                     int ncol, int ldk, int k0,
                                                     int hi) {
  const __bf16* p = W + (size_t)ncol * ldk + k0 + hi * 8;
  return combine16(*(const v8bf*)(p), *(const v8bf*)(p + 16));
}

__device__ __forceinline__ float warp_sum32(float v) {
#pragma unroll
  for (int m = 1; m < 32; m <<= 1) v += __shfl_xor(v, m, 32);
  return v;
}

// ---------------------------------------------------------------------------
// CDNA5 data movers
// ---------------------------------------------------------------------------
// Per-lane async copy of one 16-byte chunk: global (sbase + voff) -> LDS.
__device__ __forceinline__ void async_b128(unsigned lds_addr, unsigned voff,
                                           const void* sbase) {
  asm volatile("global_load_async_to_lds_b128 %0, %1, %2"
               :: "v"(lds_addr), "v"(voff), "s"(sbase) : "memory");
}
__device__ __forceinline__ void wait_async0() {
  asm volatile("s_wait_asynccnt 0x0" ::: "memory");
}
__device__ __forceinline__ void wait_ds0() {
  asm volatile("s_wait_dscnt 0x0" ::: "memory");
}

// Tensor Data Mover: 2D tile load  Global[rows x tile_cols, row stride = K
// elements, 2-byte elements] -> LDS[lds_addr].  One instruction per wave,
// tracked with TENSORcnt.  Descriptors per CDNA5 ISA.
__device__ __forceinline__ void tdm_load_tile_2d(unsigned lds_addr, const void* gaddr,
                                                 unsigned K, unsigned rows,
                                                 unsigned tile_cols) {
  unsigned long long ga = (unsigned long long)(uintptr_t)gaddr;
  u32x4 g0;
  g0[0] = 1u;                                            // count=1, user D#
  g0[1] = lds_addr;                                      // lds_addr (bytes)
  g0[2] = (unsigned)(ga & 0xFFFFFFFFu);                  // global_addr[31:0]
  g0[3] = (unsigned)((ga >> 32) & 0x1FFFFFFu) | (2u << 30);  // addr[56:32]|type=2
  u32x8 g1;
  g1[0] = 1u << 16;                                      // data_size=1 (2 B)
  g1[1] = (K & 0xFFFFu) << 16;                           // tensor_dim0 lo16
  g1[2] = (K >> 16) | ((rows & 0xFFFFu) << 16);          // dim0 hi | dim1 lo
  g1[3] = (rows >> 16) | ((tile_cols & 0xFFFFu) << 16);  // dim1 hi | tile_dim0
  g1[4] = rows & 0xFFFFu;                                // tile_dim1 (tile_dim2=0)
  g1[5] = K;                                             // dim0_stride lo32
  g1[6] = 0u;                                            // stride hi | dim1_stride
  g1[7] = 0u;
  u32x4 gz = {0u, 0u, 0u, 0u};                           // zeroed groups 2/3 (2D)
  asm volatile("tensor_load_to_lds %0, %1, %2, %3"
               :: "s"(g0), "s"(g1), "s"(gz), "s"(gz) : "memory");
}
__device__ __forceinline__ void wait_tensor0() {
  __builtin_amdgcn_s_wait_tensorcnt((short)0);
}

// ---------------------------------------------------------------------------
// f32 -> bf16 packing kernels
// ---------------------------------------------------------------------------
__global__ void k_pack(const float* __restrict__ src, __bf16* __restrict__ dst, int n) {
  int i = blockIdx.x * 256 + threadIdx.x;
  if (i < n) dst[i] = (__bf16)src[i];
}

// conv_w OIHW [256,512,3,3] -> bf16 [tap=ky*3+kx][o][i]
__global__ void k_pack_conv(const float* __restrict__ src, __bf16* __restrict__ dst) {
  int i = blockIdx.x * 256 + threadIdx.x;
  if (i < 256 * 512 * 9) {
    int tap = i % 9;
    int ic  = (i / 9) % 512;
    int oc  = i / (9 * 512);
    dst[((size_t)tap * 256 + oc) * 512 + ic] = (__bf16)src[i];
  }
}

// ---------------------------------------------------------------------------
// WMMA GEMM:  C[M,N] = A[M,K] * W[N,K]^T + bias
// EPI: 0 = f32 out; 1 = bf16 out; 2 = bf16 out with exact GELU.
// Block = 128 threads (4 waves), wave tile 16(M)x64(N); all waves share n0 so
// the 64-row weight tile is staged in LDS by the TENSOR DATA MOVER (one
// tensor_load_to_lds per K-stage, issued by wave 0, TENSORcnt-fenced).
// Grid (M/64, N/64), dyn-shared = 32768 B.
// ---------------------------------------------------------------------------
#define KSTAGE 256

template <int EPI>
__global__ void k_gemm(const __bf16* __restrict__ A, const __bf16* __restrict__ W,
                       const float* __restrict__ bias, void* __restrict__ out,
                       int M, int N, int K) {
  extern __shared__ __bf16 ldsW[];  // [64][KSTAGE]
  const int tid  = threadIdx.x;
  const int lane = tid & 31;
  const int wave = tid >> 5;
  const int l15  = lane & 15;
  const int hi   = lane >> 4;
  const int m0   = (blockIdx.x * 4 + wave) * 16;
  const int n0   = blockIdx.y * 64;
  const int mrow = m0 + l15;

  const unsigned ldsbase = (unsigned)(uintptr_t)(void*)ldsW;
  const __bf16* Wt = W + (size_t)n0 * K;  // 64 rows, row stride K elements

  v8f acc[4] = {};
  for (int ks = 0; ks < K; ks += KSTAGE) {
    // ---- TDM: stage W[n0..n0+64, ks..ks+KSTAGE) into LDS -------------------
    if (wave == 0) {
      tdm_load_tile_2d(ldsbase, (const void*)((const char*)Wt + (size_t)ks * 2),
                       (unsigned)K, 64u, (unsigned)KSTAGE);
      wait_tensor0();
    }
    __syncthreads();

    // ---- MACs over this K stage -------------------------------------------
    for (int k0 = 0; k0 < KSTAGE; k0 += 32) {
      v16bf a = load_afrag(A, mrow, K, ks + k0, l15, hi);
      __builtin_prefetch(A + (size_t)mrow * K + ks + k0 + 32, 0, 1);
#pragma unroll
      for (int j = 0; j < 4; ++j) {
        v16bf b = load_bfrag_rowmajor(ldsW, j * 16 + l15, KSTAGE, k0, hi);
        acc[j] = wmma_bf16(a, b, acc[j]);
      }
    }
    __syncthreads();  // before next stage overwrites the tile
  }

  // ---- epilogue: lane col = n0+j*16+l15, VGPR r -> row m0+r+8*hi ----------
#pragma unroll
  for (int j = 0; j < 4; ++j) {
    int ncol = n0 + j * 16 + l15;
    float bv = bias[ncol];
#pragma unroll
    for (int r = 0; r < 8; ++r) {
      int row = m0 + r + hi * 8;
      float v = acc[j][r] + bv;
      if (EPI == 2) v = 0.5f * v * (1.0f + erff(v * 0.70710678118654752f));
      if (EPI == 0) ((float*)out)[(size_t)row * N + ncol] = v;
      else          ((__bf16*)out)[(size_t)row * N + ncol] = (__bf16)v;
    }
  }
}

// ---------------------------------------------------------------------------
// Flash cross-attention (one direction).  Pq/Pkv: bf16 [4096,512]
// ([l, f*32+d]=qk, [l, 256+f*32+d]=v);  Out: bf16 [4096,256].
// One wave = one (head, 16-row l-tile).  Scores bounce through LDS so each
// lane holds one row's 16 values in A-fragment order (one xor-16 shuffle per
// reduction, P fragment built in registers).  V chunks are staged into LDS
// with per-lane async-to-LDS copies (ASYNCcnt-fenced, no VGPR staging).
// ---------------------------------------------------------------------------
__global__ void k_attn(const __bf16* __restrict__ Pq, const __bf16* __restrict__ Pkv,
                       __bf16* __restrict__ Out) {
  __shared__ __attribute__((aligned(16))) float  sc[4][16 * 32];   // score tile
  __shared__ __attribute__((aligned(16))) __bf16 vb[4][32 * 32];   // V chunk
  __shared__ __attribute__((aligned(16))) float  esb[4][16];       // rescale
  __shared__ __attribute__((aligned(16))) float  invb[4][16];      // 1/l

  const int lane = threadIdx.x & 31;
  const int wave = threadIdx.x >> 5;
  const int l15  = lane & 15;
  const int hi   = lane >> 4;

  const int wid   = blockIdx.x * 4 + wave;  // 0..2047
  const int f     = wid >> 8;               // head
  const int lbase = (wid & 255) * 16;       // l-tile base

  const unsigned vbbase = (unsigned)(uintptr_t)(void*)&vb[wave][0];

  // Q fragment (row l = lbase+l15, contiguous 32-element head-dim)
  v16bf q;
  {
    const __bf16* pq = Pq + (size_t)(lbase + l15) * 512 + f * 32 + hi * 8;
    q = combine16(*(const v8bf*)pq, *(const v8bf*)(pq + 16));
  }

  v8f acc0 = {}, acc1 = {};
  float mrow = -1e30f, lrow = 0.0f;          // this lane's row stats (row = l15)
  const float scale = 0.17677669529663687f;  // 1/sqrt(32)

  for (int sb = 0; sb < 4096; sb += 32) {
    // ---- async-stage V rows straight into LDS (no VGPR staging) -----------
    {
      unsigned voff = (unsigned)((sb + lane) * 1024 + (256 + f * 32) * 2);
      unsigned loff = vbbase + (unsigned)(lane * 64);
#pragma unroll
      for (int j2 = 0; j2 < 4; ++j2)
        async_b128(loff + j2 * 16, voff + j2 * 16, (const void*)Pkv);
    }

    // ---- K fragments + score WMMAs ----------------------------------------
    v16bf kf0, kf1;
    {
      const __bf16* pk0 = Pkv + (size_t)(sb + l15) * 512 + f * 32 + hi * 8;
      kf0 = combine16(*(const v8bf*)pk0, *(const v8bf*)(pk0 + 16));
      const __bf16* pk1 = Pkv + (size_t)(sb + 16 + l15) * 512 + f * 32 + hi * 8;
      kf1 = combine16(*(const v8bf*)pk1, *(const v8bf*)(pk1 + 16));
    }
    v8f s0 = {}, s1 = {};
    s0 = wmma_bf16(q, kf0, s0);
    s1 = wmma_bf16(q, kf1, s1);

    // ---- scores -> LDS (C-layout write) -----------------------------------
#pragma unroll
    for (int r = 0; r < 8; ++r) {
      sc[wave][(r + 8 * hi) * 32 + l15]      = s0[r] * scale;
      sc[wave][(r + 8 * hi) * 32 + 16 + l15] = s1[r] * scale;
    }
    wait_ds0();

    // ---- read back row l15 in A-fragment element order --------------------
    const float* srow = &sc[wave][l15 * 32 + hi * 8];
    v4f a0 = *(const v4f*)(srow);
    v4f a1 = *(const v4f*)(srow + 4);
    v4f a2 = *(const v4f*)(srow + 16);
    v4f a3 = *(const v4f*)(srow + 20);
    float vv[16];
#pragma unroll
    for (int e = 0; e < 4; ++e) {
      vv[e] = a0[e]; vv[4 + e] = a1[e]; vv[8 + e] = a2[e]; vv[12 + e] = a3[e];
    }
    float rm = vv[0];
#pragma unroll
    for (int e = 1; e < 16; ++e) rm = fmaxf(rm, vv[e]);
    rm = fmaxf(rm, __shfl_xor(rm, 16, 32));   // combine both column halves
    float nm = fmaxf(mrow, rm);
    float es = __expf(mrow - nm);
    mrow = nm;
    float p[16], rs = 0.0f;
#pragma unroll
    for (int e = 0; e < 16; ++e) { p[e] = __expf(vv[e] - nm); rs += p[e]; }
    rs += __shfl_xor(rs, 16, 32);
    lrow = lrow * es + rs;
    if (hi == 0) esb[wave][l15] = es;

    // P fragment directly in registers (already A-layout ordered)
    v16bf pf;
#pragma unroll
    for (int e = 0; e < 16; ++e) pf[e] = (__bf16)p[e];
    wait_ds0();

    // ---- rescale accumulators (C-layout rows r+8*hi) ----------------------
    const float* ep = &esb[wave][8 * hi];
#pragma unroll
    for (int r = 0; r < 8; ++r) {
      float e = ep[r];
      acc0[r] *= e;
      acc1[r] *= e;
    }

    // ---- V B-fragments from LDS + PV WMMAs --------------------------------
    wait_async0();   // V chunk resident in LDS
    v16bf vf0, vf1;
#pragma unroll
    for (int e = 0; e < 16; ++e) {
      vf0[e] = vb[wave][(e < 8 ? 8 * hi + e : 8 + 8 * hi + e) * 32 + l15];
      vf1[e] = vb[wave][(e < 8 ? 8 * hi + e : 8 + 8 * hi + e) * 32 + 16 + l15];
    }
    acc0 = wmma_bf16(pf, vf0, acc0);
    acc1 = wmma_bf16(pf, vf1, acc1);
  }

  // ---- normalize + store ---------------------------------------------------
  if (hi == 0) invb[wave][l15] = 1.0f / lrow;
  wait_ds0();
  const float* ip = &invb[wave][8 * hi];
#pragma unroll
  for (int r = 0; r < 8; ++r) {
    int l = lbase + r + hi * 8;
    float inv = ip[r];
    Out[(size_t)l * 256 + f * 32 + l15]      = (__bf16)(acc0[r] * inv);
    Out[(size_t)l * 256 + f * 32 + 16 + l15] = (__bf16)(acc1[r] * inv);
  }
}

// ---------------------------------------------------------------------------
// LayerNorm(Mf, norm0) and concat with bf16(x) into Y[8192,512].
// ---------------------------------------------------------------------------
__global__ void k_concat_ln(const float* __restrict__ Mf,
                            const float* __restrict__ x0, const float* __restrict__ x1,
                            const float* __restrict__ g, const float* __restrict__ b,
                            __bf16* __restrict__ Y) {
  __shared__ float s1[8], s2[8];
  int row = blockIdx.x, c = threadIdx.x;
  const float* xs = (row < 4096) ? x0 : x1;
  float xv = xs[(size_t)(row & 4095) * 256 + c];
  float v  = Mf[(size_t)row * 256 + c];
  float a  = warp_sum32(v);
  float sq = warp_sum32(v * v);
  int lane = c & 31, wid = c >> 5;
  if (lane == 0) { s1[wid] = a; s2[wid] = sq; }
  __syncthreads();
  float sum = 0.0f, ss = 0.0f;
#pragma unroll
  for (int i = 0; i < 8; ++i) { sum += s1[i]; ss += s2[i]; }
  float mu  = sum * (1.0f / 256.0f);
  float var = ss * (1.0f / 256.0f) - mu * mu;
  float rs  = rsqrtf(var + 1e-5f);
  float ln  = (v - mu) * rs * g[c] + b[c];
  Y[(size_t)row * 512 + c]       = (__bf16)xv;
  Y[(size_t)row * 512 + 256 + c] = (__bf16)ln;
}

// ---------------------------------------------------------------------------
// 3x3 conv (SAME) as implicit GEMM over 9 taps.
// G: bf16 [2*4096,512]; Wp: bf16 [9][256][512].  Block=128 (4 waves x N64),
// one block per (img, y, 16-wide x-tile).  Grid = 512.
// ---------------------------------------------------------------------------
__global__ void k_conv3x3(const __bf16* __restrict__ G, const __bf16* __restrict__ Wp,
                          const float* __restrict__ bias, float* __restrict__ out) {
  const int lane = threadIdx.x & 31;
  const int wave = threadIdx.x >> 5;
  const int l15  = lane & 15;
  const int hi   = lane >> 4;
  const int bidx = blockIdx.x;
  const int img  = bidx >> 8;
  const int y    = (bidx & 255) >> 2;
  const int xt   = bidx & 3;
  const int x    = xt * 16 + l15;
  const int n0   = wave * 64;

  v8f acc[4] = {};
#pragma unroll
  for (int tap = 0; tap < 9; ++tap) {
    const int dy = tap / 3 - 1, dx = tap % 3 - 1;
    const int yy = y + dy, xx = x + dx;
    const bool valid = (yy >= 0) && (yy < 64) && (xx >= 0) && (xx < 64);
    const __bf16* rowp = G + ((size_t)(img * 4096 + yy * 64 + xx)) * 512;
    const __bf16* wp   = Wp + (size_t)tap * 256 * 512;
    for (int k0 = 0; k0 < 512; k0 += 32) {
      v16bf a;
      if (valid) {
        const __bf16* p = rowp + k0 + hi * 8;
        a = combine16(*(const v8bf*)p, *(const v8bf*)(p + 16));
      } else {
#pragma unroll
        for (int i = 0; i < 16; ++i) a[i] = (__bf16)0.0f;
      }
#pragma unroll
      for (int j = 0; j < 4; ++j) {
        v16bf b = load_bfrag_rowmajor(wp, n0 + j * 16 + l15, 512, k0, hi);
        acc[j] = wmma_bf16(a, b, acc[j]);
      }
    }
  }
#pragma unroll
  for (int j = 0; j < 4; ++j) {
    int ncol = n0 + j * 16 + l15;
    float bv = bias[ncol];
#pragma unroll
    for (int r = 0; r < 8; ++r) {
      int xo = xt * 16 + r + hi * 8;
      size_t row = (size_t)(img * 4096 + y * 64 + xo);
      out[row * 256 + ncol] = acc[j][r] + bv;
    }
  }
}

// ---------------------------------------------------------------------------
// Final LayerNorm(norm1) + residual -> d_out (f32, y0 flat then y1 flat).
// ---------------------------------------------------------------------------
__global__ void k_final_ln(const float* __restrict__ C,
                           const float* __restrict__ x0, const float* __restrict__ x1,
                           const float* __restrict__ g, const float* __restrict__ b,
                           float* __restrict__ out) {
  __shared__ float s1[8], s2[8];
  int row = blockIdx.x, c = threadIdx.x;
  float v  = C[(size_t)row * 256 + c];
  float a  = warp_sum32(v);
  float sq = warp_sum32(v * v);
  int lane = c & 31, wid = c >> 5;
  if (lane == 0) { s1[wid] = a; s2[wid] = sq; }
  __syncthreads();
  float sum = 0.0f, ss = 0.0f;
#pragma unroll
  for (int i = 0; i < 8; ++i) { sum += s1[i]; ss += s2[i]; }
  float mu  = sum * (1.0f / 256.0f);
  float var = ss * (1.0f / 256.0f) - mu * mu;
  float rs  = rsqrtf(var + 1e-5f);
  const float* xs = (row < 4096) ? x0 : x1;
  float ln = (v - mu) * rs * g[c] + b[c];
  out[(size_t)row * 256 + c] = ln + xs[(size_t)(row & 4095) * 256 + c];
}

// ---------------------------------------------------------------------------
// Host orchestration
// ---------------------------------------------------------------------------
extern "C" void kernel_launch(void* const* d_in, const int* in_sizes, int n_in,
                              void* d_out, int out_size, void* d_ws, size_t ws_size,
                              hipStream_t stream) {
  (void)in_sizes; (void)n_in; (void)out_size; (void)ws_size;
  const float* x0      = (const float*)d_in[0];
  const float* x1      = (const float*)d_in[1];
  // d_in[2..5] = h0,w0,h1,w1 (64x64, hardcoded)
  const float* proj_w  = (const float*)d_in[6];
  const float* proj_b  = (const float*)d_in[7];
  const float* merge_w = (const float*)d_in[8];
  const float* merge_b = (const float*)d_in[9];
  const float* n0g     = (const float*)d_in[10];
  const float* n0b     = (const float*)d_in[11];
  const float* lin_w   = (const float*)d_in[12];
  const float* lin_b   = (const float*)d_in[13];
  const float* conv_w  = (const float*)d_in[14];
  const float* conv_b  = (const float*)d_in[15];
  const float* n1g     = (const float*)d_in[16];
  const float* n1b     = (const float*)d_in[17];

  char* ws = (char*)d_ws;
  __bf16* wproj  = (__bf16*)(ws);                         // 512*256
  __bf16* wmerge = wproj  + 512 * 256;                    // 256*256
  __bf16* wlin   = wmerge + 256 * 256;                    // 512*512
  __bf16* wconv  = wlin   + 512 * 512;                    // 9*256*512
  char* slotA = ws + 3276800;              // 4 MB : Xb (x bf16)  -> Mh (attn out)
  char* slotB = slotA + 8192 * 256 * 2;    // 8 MB : Pb (proj out) -> Y (concat)
  char* slotC = slotB + 8192 * 512 * 2;    // 8 MB : Mf (merge f32) -> G (gelu bf16)
  char* slotD = slotC + 8192 * 512 * 2;    // 8 MB : conv out f32
  __bf16* Xb   = (__bf16*)slotA;
  __bf16* Mh   = (__bf16*)slotA;
  __bf16* Pb   = (__bf16*)slotB;
  __bf16* Y    = (__bf16*)slotB;
  float*  Mf   = (float*)slotC;
  __bf16* G    = (__bf16*)slotC;
  float*  Cbuf = (float*)slotD;

  const size_t gemm_lds = 64 * KSTAGE * 2;  // 32 KB weight tile

  // 1) pack inputs + weights to bf16
  k_pack<<<4096, 256, 0, stream>>>(x0, Xb, 4096 * 256);
  k_pack<<<4096, 256, 0, stream>>>(x1, Xb + 4096 * 256, 4096 * 256);
  k_pack<<<512,  256, 0, stream>>>(proj_w,  wproj,  512 * 256);
  k_pack<<<256,  256, 0, stream>>>(merge_w, wmerge, 256 * 256);
  k_pack<<<1024, 256, 0, stream>>>(lin_w,   wlin,   512 * 512);
  k_pack_conv<<<4608, 256, 0, stream>>>(conv_w, wconv);

  // 2) proj: P[8192,512] = X @ proj_w^T + proj_b (bf16 out)
  k_gemm<1><<<dim3(128, 8), 128, gemm_lds, stream>>>(Xb, wproj, proj_b, Pb,
                                                     8192, 512, 256);

  // 3) bidirectional cross-attention (two flash passes, Q/K swapped)
  k_attn<<<512, 128, 0, stream>>>(Pb, Pb + (size_t)4096 * 512, Mh);
  k_attn<<<512, 128, 0, stream>>>(Pb + (size_t)4096 * 512, Pb, Mh + (size_t)4096 * 256);

  // 4) merge: Mf[8192,256] = Mh @ merge_w^T + merge_b (f32 out)
  k_gemm<0><<<dim3(128, 4), 128, gemm_lds, stream>>>(Mh, wmerge, merge_b, Mf,
                                                     8192, 256, 256);

  // 5) LN(norm0) + concat -> Y[8192,512] bf16
  k_concat_ln<<<8192, 256, 0, stream>>>(Mf, x0, x1, n0g, n0b, Y);

  // 6) lin + GELU: G[8192,512] (bf16)
  k_gemm<2><<<dim3(128, 8), 128, gemm_lds, stream>>>(Y, wlin, lin_b, G,
                                                     8192, 512, 512);

  // 7) 3x3 conv (implicit GEMM) -> Cbuf f32
  k_conv3x3<<<512, 128, 0, stream>>>(G, wconv, conv_b, Cbuf);

  // 8) LN(norm1) + residual -> d_out
  k_final_ln<<<8192, 256, 0, stream>>>(Cbuf, x0, x1, n1g, n1b, (float*)d_out);
}
